// GraphConvolution_5531917877260
// MI455X (gfx1250) — matrix operand order repaired and checked by reading, not compiled, not run on hardware
//
#include <hip/hip_runtime.h>

typedef __attribute__((ext_vector_type(16))) _Float16 v16h;
typedef __attribute__((ext_vector_type(8)))  float    v8f;

#define BB 4
#define NN 50000
#define FF 64        // F_IN == F_OUT == 64
#define KK 4
#define EE 800000
#define ROWS (BB * NN)   // 200000 rows of 64 features

// ---------------------------------------------------------------------------
// GEMM: H[row, 0:64] = x[row, 0:64] @ Wk[64, 64]  via v_wmma_f32_16x16x32_f16
// One wave computes a 16(row) x 64(col) tile.
// Wk is pre-swizzled into LDS in B-fragment order: sB[tile][lane] is the v16h
// that lane needs for that WMMA -> each B fragment = 2x ds_load_b128.
//   tile = s*4 + t  (s = f-step 0..1, t = g-tile 0..3)
//   lane l holds B[k][g] with k = s*32 + (l>>4)*16 + j (j=0..15), g = t*16+(l&15)
// ---------------------------------------------------------------------------
__global__ __launch_bounds__(256)
void gcn_gemm_wmma(const float* __restrict__ x,
                   const float* __restrict__ Wk,   // [64 (f), 64 (g)] row-major
                   float* __restrict__ H,
                   int total_rows)
{
    __shared__ v16h sB[8 * 32];     // 8 KB: 8 tiles x 32 lanes x 16 halves

    // Cooperative swizzle-fill: 256 threads, one v16h chunk each.
    {
        const int tid  = threadIdx.x;
        const int tile = tid >> 5;          // 0..7
        const int l    = tid & 31;
        const int s    = tile >> 2;
        const int t    = tile & 3;
        const int kb   = s * 32 + (l >> 4) * 16;
        const int g    = t * 16 + (l & 15);
        v16h w;
        #pragma unroll
        for (int j = 0; j < 16; ++j)
            w[j] = (_Float16)Wk[(kb + j) * FF + g];
        sB[tid] = w;
    }
    __syncthreads();

    const int wave = (int)blockIdx.x * (blockDim.x >> 5) + ((int)threadIdx.x >> 5);
    const int lane = threadIdx.x & 31;
    const int row0 = wave * 16;
    if (row0 >= total_rows) return;

    const int m       = lane & 15;   // row within tile (A) / col within tile (C,D)
    const int half_hi = lane >> 4;   // lane group -> which K-slice this lane holds

    v8f acc0 = {}, acc1 = {}, acc2 = {}, acc3 = {};

    const float4* xv = (const float4*)(x + (size_t)(row0 + m) * FF);

    #pragma unroll
    for (int s = 0; s < 2; ++s) {          // K = 64 in two steps of 32
        // A fragment (16x32 f16): lane group 0 holds K j=0..15 of this step,
        // group 1 holds 16..31 (matches sB order).
        const int q = s * 8 + half_hi * 4;            // float4 index base
        const float4 a0 = xv[q + 0];
        const float4 a1 = xv[q + 1];
        const float4 a2 = xv[q + 2];
        const float4 a3 = xv[q + 3];
        v16h a;
        a[0]  = (_Float16)a0.x; a[1]  = (_Float16)a0.y;
        a[2]  = (_Float16)a0.z; a[3]  = (_Float16)a0.w;
        a[4]  = (_Float16)a1.x; a[5]  = (_Float16)a1.y;
        a[6]  = (_Float16)a1.z; a[7]  = (_Float16)a1.w;
        a[8]  = (_Float16)a2.x; a[9]  = (_Float16)a2.y;
        a[10] = (_Float16)a2.z; a[11] = (_Float16)a2.w;
        a[12] = (_Float16)a3.x; a[13] = (_Float16)a3.y;
        a[14] = (_Float16)a3.z; a[15] = (_Float16)a3.w;

        #pragma unroll
        for (int t = 0; t < 4; ++t) {      // four 16-wide g tiles -> g = 0..63
            const v16h b = sB[(s * 4 + t) * 32 + lane];   // 2x ds_load_b128
            v8f* accp = (t == 0) ? &acc0 : (t == 1) ? &acc1 : (t == 2) ? &acc2 : &acc3;
            *accp = __builtin_amdgcn_wmma_f32_16x16x32_f16(
                false, a, false, b, (short)0, *accp, false, false);
        }
    }

    // D layout: VGPR r, lanes 0-15 -> M=r, lanes 16-31 -> M=r+8; N = lane&15.
    #pragma unroll
    for (int r = 0; r < 8; ++r) {
        float* hp = H + (size_t)(row0 + r + 8 * half_hi) * FF + m;
        hp[0]  = acc0[r];
        hp[16] = acc1[r];
        hp[32] = acc2[r];
        hp[48] = acc3[r];
    }
}

// ---------------------------------------------------------------------------
// Scatter: out[b, row, :] += val * H[b, col, :]   (64 threads per edge,
// one float4 gather + 4 native f32 atomics each). Bandwidth-dominant phase:
// ~6.6 GB total -> ~285 us at 23.3 TB/s.
// ---------------------------------------------------------------------------
__global__ __launch_bounds__(256)
void gcn_scatter(const float* __restrict__ H,
                 const int*   __restrict__ rows,
                 const int*   __restrict__ cols,
                 const float* __restrict__ vals,
                 float*       __restrict__ out,
                 int num_edges)
{
    const int t = (int)blockIdx.x * blockDim.x + (int)threadIdx.x;
    const int e = t >> 6;
    if (e >= num_edges) return;
    const int sub = t & 63;
    const int b   = sub >> 4;          // 0..3
    const int g   = (sub & 15) * 4;    // 0,4,...,60

    const int   r = rows[e];
    const int   c = cols[e];
    const float v = vals[e];

    const float4 h4 = *(const float4*)(H + ((size_t)b * NN + c) * FF + g);
    float* dst = out + ((size_t)b * NN + r) * FF + g;
    atomicAdd(dst + 0, v * h4.x);
    atomicAdd(dst + 1, v * h4.y);
    atomicAdd(dst + 2, v * h4.z);
    atomicAdd(dst + 3, v * h4.w);
}

// ---------------------------------------------------------------------------
__global__ __launch_bounds__(256)
void gcn_zero(float4* __restrict__ out, int n4)
{
    const int i = (int)blockIdx.x * blockDim.x + (int)threadIdx.x;
    if (i < n4) out[i] = make_float4(0.f, 0.f, 0.f, 0.f);
}

__global__ __launch_bounds__(256)
void gcn_bias_relu(float4* __restrict__ out, const float* __restrict__ bias, int n4)
{
    const int i = (int)blockIdx.x * blockDim.x + (int)threadIdx.x;
    if (i >= n4) return;
    const float4 bv = ((const float4*)bias)[i & 15];   // 64 feats = 16 float4s
    float4 v = out[i];
    v.x += bv.x; v.y += bv.y; v.z += bv.z; v.w += bv.w;
    v.x = v.x > 0.f ? v.x : 0.f;
    v.y = v.y > 0.f ? v.y : 0.f;
    v.z = v.z > 0.f ? v.z : 0.f;
    v.w = v.w > 0.f ? v.w : 0.f;
    out[i] = v;
}

// ---------------------------------------------------------------------------
extern "C" void kernel_launch(void* const* d_in, const int* in_sizes, int n_in,
                              void* d_out, int out_size, void* d_ws, size_t ws_size,
                              hipStream_t stream)
{
    const float* x     = (const float*)d_in[0];   // [B, N, 64]
    const int*   rows  = (const int*)  d_in[1];   // [K, E]
    const int*   cols  = (const int*)  d_in[2];   // [K, E]
    const float* vals  = (const float*)d_in[3];   // [K, E]
    const float* kerns = (const float*)d_in[4];   // [K, 64, 64]
    const float* bias  = (const float*)d_in[5];   // [64]
    float* out = (float*)d_out;                   // [B, N, 64] f32
    float* H   = (float*)d_ws;                    // [B, N, 64] f32 scratch (51.2 MB)

    const int n4 = ROWS * (FF / 4);               // float4 count of out

    gcn_zero<<<(n4 + 255) / 256, 256, 0, stream>>>((float4*)out, n4);

    const int gemm_waves  = (ROWS + 15) / 16;                 // 12500
    const int gemm_blocks = (gemm_waves + 7) / 8;             // 8 waves / 256-thr block
    const int scat_blocks = (EE * 64 + 255) / 256;

    for (int k = 0; k < KK; ++k) {
        gcn_gemm_wmma<<<gemm_blocks, 256, 0, stream>>>(
            x, kerns + (size_t)k * FF * FF, H, ROWS);
        gcn_scatter<<<scat_blocks, 256, 0, stream>>>(
            H, rows + (size_t)k * EE, cols + (size_t)k * EE,
            vals + (size_t)k * EE, out, EE);
    }

    gcn_bias_relu<<<(n4 + 255) / 256, 256, 0, stream>>>((float4*)out, bias, n4);
}
